// GraphBiasedMHA_34746285425453
// MI455X (gfx1250) — matrix operand order repaired and checked by reading, not compiled, not run on hardware
//
#include <hip/hip_runtime.h>

// ---------------------------------------------------------------------------
// Biased MHA for MI455X (gfx1250): bf16 WMMA + TDM (tensor_load_to_lds)
// staged tiles, flash-attention single pass. K-loops unrolled x2 so the
// WMMA accumulator chain alternates register sets without v_mov copies.
// ---------------------------------------------------------------------------

typedef __attribute__((ext_vector_type(16))) __bf16 v16bf;
typedef __attribute__((ext_vector_type(8)))  float  v8f;
typedef __attribute__((ext_vector_type(4)))  unsigned int v4u;
typedef __attribute__((ext_vector_type(8)))  int    v8i;
typedef __attribute__((ext_vector_type(4)))  int    v4i;

constexpr int Bb   = 4;
constexpr int Nn   = 2048;
constexpr int Dd   = 1024;
constexpr int Hh   = 16;
constexpr int DH   = 64;          // head dim
constexpr int NCTX = 512;
constexpr int Mrows = Bb * Nn;    // 8192
constexpr float SCALE = 0.125f;   // 1/sqrt(64)

#if __has_builtin(__builtin_amdgcn_tensor_load_to_lds) && \
    __has_builtin(__builtin_amdgcn_s_wait_tensorcnt)
#define USE_TDM 1
#define TDM_WAIT(n) __builtin_amdgcn_s_wait_tensorcnt(n)
#else
#define USE_TDM 0
#define TDM_WAIT(n)
#endif

// ---------------------------------------------------------------------------
// TDM 2-D tile load: stage (tile_rows x tile_k) bf16 tile (row stride
// `row_stride` elements) from global into LDS, rows compacted to tile_k.
// D# layout per CDNA5 ISA. Issued by one wave.
// ---------------------------------------------------------------------------
__device__ inline void stage_tile(__bf16* lds_dst, const __bf16* __restrict__ gsrc,
                                  int tile_k, int tile_rows,
                                  unsigned tdim0, unsigned tdim1,
                                  unsigned row_stride) {
#if USE_TDM
  const unsigned long long ga = (unsigned long long)(const void*)gsrc;
  v4u g0;
  g0[0] = 1u;                                           // count=1, user mode
  g0[1] = (unsigned)(size_t)(void*)lds_dst;             // lds_addr
  g0[2] = (unsigned)ga;                                 // global_addr[31:0]
  g0[3] = (unsigned)((ga >> 32) & 0x01ffffffu) | (2u << 30); // addr[56:32]|type=2
  v8i g1;
  g1[0] = (int)(1u << 16);                              // data_size=1 (2 bytes)
  g1[1] = (int)((tdim0 & 0xffffu) << 16);               // tensor_dim0[15:0]
  g1[2] = (int)((tdim0 >> 16) | ((tdim1 & 0xffffu) << 16));
  g1[3] = (int)((tdim1 >> 16) | ((unsigned)tile_k << 16)); // tile_dim0
  g1[4] = tile_rows;                                    // tile_dim1 (tile_dim2=0)
  g1[5] = (int)row_stride;                              // tensor_dim0_stride lo
  g1[6] = 0;
  g1[7] = 0;
  v4i z4 = {0, 0, 0, 0};
#if defined(__clang_major__) && (__clang_major__ >= 23)
  v8i z8 = {0, 0, 0, 0, 0, 0, 0, 0};
  __builtin_amdgcn_tensor_load_to_lds(g0, g1, z4, z4, z8, 0);
#else
  __builtin_amdgcn_tensor_load_to_lds(g0, g1, z4, z4, 0);
#endif
#else
  // fallback: cooperative copy by the issuing wave (32 lanes), dword granular
  const int lane = threadIdx.x & 31;
  const int dpr = tile_k >> 1;
  const int total = tile_rows * dpr;
  for (int i = lane; i < total; i += 32) {
    const int r = i / dpr, c = i - r * dpr;
    reinterpret_cast<unsigned*>(lds_dst)[i] =
        reinterpret_cast<const unsigned*>(gsrc + (size_t)r * row_stride)[c];
  }
#endif
}

// ---------------------------------------------------------------------------
// Fragment loaders (CDNA5 WMMA 16x16x32 bf16 layouts, CDNA5 ISA §7.12.2)
// ---------------------------------------------------------------------------

__device__ inline v16bf load_a_frag(const __bf16* A, int lda) {
  const int lane = threadIdx.x & 31;
  const int m    = lane & 15;
  const int half = lane >> 4;
  const __bf16* row = A + m * lda;
  v16bf a;
  unsigned int* ap = reinterpret_cast<unsigned int*>(&a);
#pragma unroll
  for (int i = 0; i < 4; ++i)
    ap[i] = *reinterpret_cast<const unsigned int*>(row + 8 * half + 2 * i);
#pragma unroll
  for (int i = 0; i < 4; ++i)
    ap[4 + i] = *reinterpret_cast<const unsigned int*>(row + 16 + 8 * half + 2 * i);
  return a;
}

__device__ inline v16bf load_b_frag(const __bf16* W, int ldb) {
  const int lane = threadIdx.x & 31;
  const int n    = lane & 15;
  const int kb   = (lane >> 4) << 3;
  const __bf16* p = W + n * ldb + kb;
  const uint4 q0 = *reinterpret_cast<const uint4*>(p);
  const uint4 q1 = *reinterpret_cast<const uint4*>(p + 16);
  v16bf b;
  unsigned int* bp = reinterpret_cast<unsigned int*>(&b);
  bp[0] = q0.x; bp[1] = q0.y; bp[2] = q0.z; bp[3] = q0.w;
  bp[4] = q1.x; bp[5] = q1.y; bp[6] = q1.z; bp[7] = q1.w;
  return b;
}

__device__ inline v8f wmma_bf16(v16bf a, v16bf b, v8f c) {
  return __builtin_amdgcn_wmma_f32_16x16x32_bf16(false, a, false, b, (short)0, c,
                                                 false, false);
}

__device__ inline v8f zero_v8f() {
  v8f z = {0.f, 0.f, 0.f, 0.f, 0.f, 0.f, 0.f, 0.f};
  return z;
}

// ---------------------------------------------------------------------------
// Prep kernels
// ---------------------------------------------------------------------------

__global__ void cvt_f32_bf16(const float* __restrict__ s, __bf16* __restrict__ d, int n) {
  for (int i = blockIdx.x * blockDim.x + threadIdx.x; i < n; i += blockDim.x * gridDim.x)
    d[i] = (__bf16)s[i];
}

__global__ void prep_gate_bias(const float* __restrict__ ppr,
                               const float* __restrict__ trust,
                               const float* __restrict__ lalpha,
                               const float* __restrict__ tscale,
                               float* __restrict__ gate_o,
                               float* __restrict__ bias_o) {
  int i = blockIdx.x * blockDim.x + threadIdx.x;
  if (i >= Nn) return;
  if (i < NCTX) {
    gate_o[i] = 1.0f / (1.0f + __expf(-tscale[0] * trust[i]));
    bias_o[i] = -lalpha[0] * __logf(fmaxf(ppr[i], 1e-8f));
  } else {
    gate_o[i] = 1.0f;
    bias_o[i] = 0.0f;
  }
}

// ---------------------------------------------------------------------------
// QKV GEMM: qkv[m,e] = sum_k x[m,k]*W_in[e,k] + b_in[e]
// Block = 4 waves -> 128(M) x 64(E); wave -> 32(M) x 64(E) (8 WMMA / k-step).
// W tile (64 x 32 bf16) staged via TDM into a double-buffered LDS ring;
// loop unrolled x2 (ring slot is compile-time).
// ---------------------------------------------------------------------------

__global__ void __launch_bounds__(128, 1)
qkv_gemm(const __bf16* __restrict__ xb,
         const __bf16* __restrict__ winb,
         const float*  __restrict__ b_in,
         const float*  __restrict__ gate,
         __bf16* __restrict__ qb,
         __bf16* __restrict__ kbuf,
         __bf16* __restrict__ vt) {
  __shared__ __bf16 bstage[2][64 * 32];

  const int wave = threadIdx.x >> 5;
  const int lane = threadIdx.x & 31;
  const int half = lane >> 4;
  const int m0   = blockIdx.x * 128 + wave * 32;
  const int n0   = blockIdx.y * 64;

  v8f acc[2][4];
#pragma unroll
  for (int i = 0; i < 2; ++i)
#pragma unroll
    for (int j = 0; j < 4; ++j) acc[i][j] = zero_v8f();

  constexpr int KT = Dd / 32;   // 32 k-steps (even)
  const __bf16* aBase = xb + (size_t)m0 * Dd;
  const __bf16* wBase = winb + (size_t)n0 * Dd;

  auto compute = [&](int kt, const __bf16* buf) {
    const int k0 = kt * 32;
    __builtin_prefetch(aBase + (lane & 15) * Dd + k0 + 64, 0, 3);
    v16bf a0 = load_a_frag(aBase + k0, Dd);
    v16bf a1 = load_a_frag(aBase + 16 * Dd + k0, Dd);
#pragma unroll
    for (int j = 0; j < 4; ++j) {
      v16bf b = load_b_frag(buf + 16 * j * 32, 32);
      acc[0][j] = wmma_bf16(a0, b, acc[0][j]);
      acc[1][j] = wmma_bf16(a1, b, acc[1][j]);
    }
  };
  auto prefetch_tile = [&](int kt_next, __bf16* buf) {
    if (wave == 0) {
      if (kt_next < KT) {
        stage_tile(buf, wBase + kt_next * 32, 32, 64, Dd, 3 * Dd, Dd);
        TDM_WAIT(1);
      } else {
        TDM_WAIT(0);
      }
    }
  };

  if (wave == 0) stage_tile(&bstage[0][0], wBase, 32, 64, Dd, 3 * Dd, Dd);

  for (int kt = 0; kt < KT; kt += 2) {
    prefetch_tile(kt + 1, &bstage[1][0]);
    __syncthreads();
    compute(kt, &bstage[0][0]);
    __syncthreads();
    prefetch_tile(kt + 2, &bstage[0][0]);
    __syncthreads();
    compute(kt + 1, &bstage[1][0]);
    __syncthreads();
  }

#pragma unroll
  for (int i = 0; i < 2; ++i) {
#pragma unroll
    for (int j = 0; j < 4; ++j) {
#pragma unroll
      for (int r = 0; r < 8; ++r) {
        const int m = m0 + 16 * i + r + 8 * half;
        const int e = n0 + 16 * j + (lane & 15);
        const int bidx = m >> 11;
        const int nseq = m & 2047;
        float val = acc[i][j][r] + b_in[e];
        if (e < Dd) {                             // Q
          const int h = e >> 6, dcol = e & 63;
          qb[(((size_t)(bidx * Hh + h)) * Nn + nseq) * DH + dcol] = (__bf16)val;
        } else if (e < 2 * Dd) {                  // K
          const int ee = e - Dd;
          const int h = ee >> 6, dcol = ee & 63;
          kbuf[(((size_t)(bidx * Hh + h)) * Nn + nseq) * DH + dcol] = (__bf16)val;
        } else {                                  // V -> gated, transposed [h,d,n]
          const int ee = e - 2 * Dd;
          const int h = ee >> 6, dcol = ee & 63;
          val *= gate[nseq];
          vt[(((size_t)(bidx * Hh + h)) * DH + dcol) * Nn + nseq] = (__bf16)val;
        }
      }
    }
  }
}

// ---------------------------------------------------------------------------
// Flash attention. Block = 4 waves = 64 queries of one (b,h); K/V^T tiles for
// the 32 keys of each step are shared by all waves -> staged via TDM ring.
// Key loop unrolled x2 (ring slot compile-time).
// ---------------------------------------------------------------------------

__global__ void __launch_bounds__(128, 1)
flash_attn(const __bf16* __restrict__ qb,
           const __bf16* __restrict__ kbuf,
           const __bf16* __restrict__ vt,
           const float*  __restrict__ bias,
           __bf16* __restrict__ attn) {
  __shared__ __bf16 kst[2][32 * 64];   // keys x d
  __shared__ __bf16 vst[2][64 * 32];   // d x keys
  __shared__ __bf16 pshare[4][16 * 32];

  const int wave = threadIdx.x >> 5;
  const int lane = threadIdx.x & 31;
  const int half = lane >> 4;
  const int col  = lane & 15;
  const int bh   = blockIdx.y;
  const int q0   = blockIdx.x * 64 + wave * 16;

  const __bf16* qBase = qb   + (size_t)bh * Nn * DH;
  const __bf16* kBase = kbuf + (size_t)bh * Nn * DH;
  const __bf16* vBase = vt   + (size_t)bh * DH * Nn;

  v16bf aq0 = load_a_frag(qBase + (size_t)q0 * DH +  0, DH);
  v16bf aq1 = load_a_frag(qBase + (size_t)q0 * DH + 32, DH);

  v8f o[4];
#pragma unroll
  for (int j = 0; j < 4; ++j) o[j] = zero_v8f();
  float rm[8], rl[8];
#pragma unroll
  for (int r = 0; r < 8; ++r) { rm[r] = -3.0e38f; rl[r] = 0.0f; }

  constexpr int NT = Nn / 32;   // 64 key steps (even)

  auto prefetch_kv = [&](int kt_next, __bf16* kdst, __bf16* vdst) {
    if (wave == 0) {
      if (kt_next < NT) {
        stage_tile(kdst, kBase + (size_t)kt_next * 32 * DH, 64, 32, 64, Nn, 64);
        stage_tile(vdst, vBase + (size_t)kt_next * 32, 32, 64, Nn, DH, Nn);
        TDM_WAIT(2);
      } else {
        TDM_WAIT(0);
      }
    }
  };

  auto attn_step = [&](int kt, const __bf16* kcur, const __bf16* vcur) {
    const int kb = kt * 32;
    // ---- S = Q K^T for 32 keys (two 16-key tiles) ----
    v8f s0 = zero_v8f(), s1 = zero_v8f();
    {
      v16bf b = load_b_frag(kcur + 0 * 64 +  0, 64);
      s0 = wmma_bf16(aq0, b, s0);
      b  = load_b_frag(kcur + 0 * 64 + 32, 64);
      s0 = wmma_bf16(aq1, b, s0);
      b  = load_b_frag(kcur + 16 * 64 +  0, 64);
      s1 = wmma_bf16(aq0, b, s1);
      b  = load_b_frag(kcur + 16 * 64 + 32, 64);
      s1 = wmma_bf16(aq1, b, s1);
    }
    const float cb0 = bias[kb + col];
    const float cb1 = bias[kb + 16 + col];

    // ---- online softmax over the 32 new keys ----
#pragma unroll
    for (int r = 0; r < 8; ++r) {
      float sv0 = s0[r] * SCALE + cb0;
      float sv1 = s1[r] * SCALE + cb1;
      float t = fmaxf(sv0, sv1);
#pragma unroll
      for (int msk = 1; msk < 16; msk <<= 1) t = fmaxf(t, __shfl_xor(t, msk, 32));
      const float mnew  = fmaxf(rm[r], t);
      const float alpha = __expf(rm[r] - mnew);
      const float p0 = __expf(sv0 - mnew);
      const float p1 = __expf(sv1 - mnew);
      float rs = p0 + p1;
#pragma unroll
      for (int msk = 1; msk < 16; msk <<= 1) rs += __shfl_xor(rs, msk, 32);
      rl[r] = rl[r] * alpha + rs;
      rm[r] = mnew;
#pragma unroll
      for (int j = 0; j < 4; ++j) o[j][r] *= alpha;
      const int m = r + 8 * half;
      pshare[wave][m * 32 + col]      = (__bf16)p0;
      pshare[wave][m * 32 + 16 + col] = (__bf16)p1;
    }
    __syncthreads();   // P tile visible

    // ---- O += P V : P as A-fragment, V^T rows as B-fragments (LDS) ----
    v16bf ap = load_a_frag(&pshare[wave][0], 32);
#pragma unroll
    for (int j = 0; j < 4; ++j) {
      v16bf bv = load_b_frag(vcur + 16 * j * 32, 32);
      o[j] = wmma_bf16(ap, bv, o[j]);
    }
  };

  if (wave == 0) {
    stage_tile(&kst[0][0], kBase, 64, 32, 64, Nn, 64);       // 32 keys x 64 d
    stage_tile(&vst[0][0], vBase, 32, 64, Nn, DH, Nn);       // 64 d x 32 keys
  }

  for (int kt = 0; kt < NT; kt += 2) {
    prefetch_kv(kt + 1, &kst[1][0], &vst[1][0]);
    __syncthreads();
    attn_step(kt, &kst[0][0], &vst[0][0]);
    __syncthreads();
    prefetch_kv(kt + 2, &kst[0][0], &vst[0][0]);
    __syncthreads();
    attn_step(kt + 1, &kst[1][0], &vst[1][0]);
    __syncthreads();
  }

  // ---- epilogue: normalize, store attn[b, n, h*64 + d] as bf16 ----
  const int bidx = bh >> 4;
  const int h    = bh & 15;
#pragma unroll
  for (int r = 0; r < 8; ++r) {
    const float inv = 1.0f / rl[r];
    const int m = q0 + r + 8 * half;
#pragma unroll
    for (int j = 0; j < 4; ++j) {
      attn[((size_t)bidx * Nn + m) * Dd + h * DH + 16 * j + col] =
          (__bf16)(o[j][r] * inv);
    }
  }
}

// ---------------------------------------------------------------------------
// Output GEMM: out[m,n] = sum_k attn[m,k]*W_out[n,k] + b_out[n]  (f32 out)
// Same structure as qkv_gemm (TDM-staged W tile, 32x64 per wave, unroll x2).
// ---------------------------------------------------------------------------

__global__ void __launch_bounds__(128, 1)
out_gemm(const __bf16* __restrict__ attn,
         const __bf16* __restrict__ woutb,
         const float*  __restrict__ b_out,
         float* __restrict__ out) {
  __shared__ __bf16 bstage[2][64 * 32];

  const int wave = threadIdx.x >> 5;
  const int lane = threadIdx.x & 31;
  const int half = lane >> 4;
  const int m0   = blockIdx.x * 128 + wave * 32;
  const int n0   = blockIdx.y * 64;

  v8f acc[2][4];
#pragma unroll
  for (int i = 0; i < 2; ++i)
#pragma unroll
    for (int j = 0; j < 4; ++j) acc[i][j] = zero_v8f();

  constexpr int KT = Dd / 32;   // even
  const __bf16* aBase = attn + (size_t)m0 * Dd;
  const __bf16* wBase = woutb + (size_t)n0 * Dd;

  auto compute = [&](int kt, const __bf16* buf) {
    const int k0 = kt * 32;
    __builtin_prefetch(aBase + (lane & 15) * Dd + k0 + 64, 0, 3);
    v16bf a0 = load_a_frag(aBase + k0, Dd);
    v16bf a1 = load_a_frag(aBase + 16 * Dd + k0, Dd);
#pragma unroll
    for (int j = 0; j < 4; ++j) {
      v16bf b = load_b_frag(buf + 16 * j * 32, 32);
      acc[0][j] = wmma_bf16(a0, b, acc[0][j]);
      acc[1][j] = wmma_bf16(a1, b, acc[1][j]);
    }
  };
  auto prefetch_tile = [&](int kt_next, __bf16* buf) {
    if (wave == 0) {
      if (kt_next < KT) {
        stage_tile(buf, wBase + kt_next * 32, 32, 64, Dd, Dd, Dd);
        TDM_WAIT(1);
      } else {
        TDM_WAIT(0);
      }
    }
  };

  if (wave == 0) stage_tile(&bstage[0][0], wBase, 32, 64, Dd, Dd, Dd);

  for (int kt = 0; kt < KT; kt += 2) {
    prefetch_tile(kt + 1, &bstage[1][0]);
    __syncthreads();
    compute(kt, &bstage[0][0]);
    __syncthreads();
    prefetch_tile(kt + 2, &bstage[0][0]);
    __syncthreads();
    compute(kt + 1, &bstage[1][0]);
    __syncthreads();
  }

#pragma unroll
  for (int i = 0; i < 2; ++i) {
#pragma unroll
    for (int j = 0; j < 4; ++j) {
#pragma unroll
      for (int r = 0; r < 8; ++r) {
        const int m = m0 + 16 * i + r + 8 * half;
        const int n = n0 + 16 * j + (lane & 15);
        out[(size_t)m * Dd + n] = acc[i][j][r] + b_out[n];
      }
    }
  }
}

// ---------------------------------------------------------------------------
// Launcher. Inputs: x, n_ctx, ctx_ppr, ctx_trust, W_in, b_in, W_out, b_out,
//                   log_ppr_alpha, trust_scale
// ---------------------------------------------------------------------------

extern "C" void kernel_launch(void* const* d_in, const int* in_sizes, int n_in,
                              void* d_out, int out_size, void* d_ws, size_t ws_size,
                              hipStream_t stream) {
  const float* x         = (const float*)d_in[0];
  const float* ctx_ppr   = (const float*)d_in[2];
  const float* ctx_trust = (const float*)d_in[3];
  const float* W_in      = (const float*)d_in[4];
  const float* b_in      = (const float*)d_in[5];
  const float* W_out     = (const float*)d_in[6];
  const float* b_out     = (const float*)d_in[7];
  const float* lalpha    = (const float*)d_in[8];
  const float* tscale    = (const float*)d_in[9];
  float* out = (float*)d_out;

  char* ws = (char*)d_ws;
  const size_t sz_xb    = (size_t)Mrows * Dd * 2;
  const size_t sz_winb  = (size_t)3 * Dd * Dd * 2;
  const size_t sz_woutb = (size_t)Dd * Dd * 2;
  const size_t sz_head  = (size_t)Bb * Hh * Nn * DH * 2;
  const size_t sz_attn  = (size_t)Mrows * Dd * 2;
  __bf16* xb    = (__bf16*)(ws);
  __bf16* winb  = (__bf16*)(ws + sz_xb);
  __bf16* woutb = (__bf16*)(ws + sz_xb + sz_winb);
  __bf16* qb    = (__bf16*)(ws + sz_xb + sz_winb + sz_woutb);
  __bf16* kbuf  = (__bf16*)(ws + sz_xb + sz_winb + sz_woutb + sz_head);
  __bf16* vt    = (__bf16*)(ws + sz_xb + sz_winb + sz_woutb + 2 * sz_head);
  __bf16* attn  = (__bf16*)(ws + sz_xb + sz_winb + sz_woutb + 3 * sz_head);
  float*  gate  = (float*)(ws + sz_xb + sz_winb + sz_woutb + 3 * sz_head + sz_attn);
  float*  bias  = gate + Nn;

  cvt_f32_bf16<<<2048, 256, 0, stream>>>(x, xb, Mrows * Dd);
  cvt_f32_bf16<<<1024, 256, 0, stream>>>(W_in, winb, 3 * Dd * Dd);
  cvt_f32_bf16<<<512, 256, 0, stream>>>(W_out, woutb, Dd * Dd);
  prep_gate_bias<<<(Nn + 255) / 256, 256, 0, stream>>>(ctx_ppr, ctx_trust, lalpha,
                                                       tscale, gate, bias);

  qkv_gemm<<<dim3(Mrows / 128, (3 * Dd) / 64), 128, 0, stream>>>(
      xb, winb, b_in, gate, qb, kbuf, vt);

  flash_attn<<<dim3(Nn / 64, Bb * Hh), 128, 0, stream>>>(qb, kbuf, vt, bias, attn);

  out_gemm<<<dim3(Mrows / 128, Dd / 64), 128, 0, stream>>>(attn, woutb, b_out, out);
}
